// DeeperGCN_1717986918670
// MI455X (gfx1250) — matrix operand order repaired and checked by reading, not compiled, not run on hardware
//
#include <hip/hip_runtime.h>
#include <math.h>

typedef __attribute__((ext_vector_type(16))) _Float16 v16h;
typedef __attribute__((ext_vector_type(8)))  float    v8f;

#define HDIM    64
#define LN_EPSF 1e-5f
#define EPS_MSG 1e-7f

// float atomic max via sign-split integer trick (init slot to -FLT_MAX).
__device__ __forceinline__ void atomicMaxF(float* addr, float val) {
    if (val >= 0.0f) atomicMax((int*)addr, __float_as_int(val));
    else             atomicMin((unsigned int*)addr, __float_as_uint(val));
}

// ---------------------------------------------------------------------------
// C[M,NOUT] = A[M,K] @ W[K,NOUT] + bias[NOUT] (+ resid[M,NOUT])
// f32 in/out, f16 WMMA compute. One wave per 16x16 tile; K,NOUT compile-time
// so the K loop fully unrolls (K/32 v_wmma ops) and all global accesses use
// immediate instruction offsets off per-lane base pointers.
// grid: (M/16, NOUT/64), block: 128 (4 waves -> 4 column tiles of 16).
// ---------------------------------------------------------------------------
template <int K, int NOUT, bool HAS_RESID>
__global__ void wmma_gemm_kernel(const float* __restrict__ A,
                                 const float* __restrict__ W,
                                 const float* __restrict__ bias,
                                 const float* __restrict__ resid,
                                 float* __restrict__ C)
{
    const int wave    = threadIdx.x >> 5;
    const int lane    = threadIdx.x & 31;
    const int laneLow = lane & 15;
    const int hi      = lane >> 4;          // half-wave select
    const int rowTile = blockIdx.x << 4;
    const int colTile = ((blockIdx.y << 2) + wave) << 4;
    const int row     = rowTile + laneLow;  // A row held by this lane
    const int col     = colTile + laneLow;  // B/C column held by this lane

    // Per-lane base pointers; all further addressing is compile-time offsets.
    const float* __restrict__ arow = A + (size_t)row * K + (hi << 4);
    const float* __restrict__ wcol = W + (size_t)(hi << 4) * NOUT + col;

    v8f c = {};
#pragma unroll
    for (int k0 = 0; k0 < K; k0 += 32) {
        v16h a, b;
        // A 16x32 f16, contiguous k order (this code's choice): lanes 0-15 -> K 0..15, lanes 16-31 -> K 16..31
        // 64B-aligned 16-float runs: force b128 loads.
        const float4 a0 = *(const float4*)(arow + k0);
        const float4 a1 = *(const float4*)(arow + k0 + 4);
        const float4 a2 = *(const float4*)(arow + k0 + 8);
        const float4 a3 = *(const float4*)(arow + k0 + 12);
        a[0]  = (_Float16)a0.x; a[1]  = (_Float16)a0.y;
        a[2]  = (_Float16)a0.z; a[3]  = (_Float16)a0.w;
        a[4]  = (_Float16)a1.x; a[5]  = (_Float16)a1.y;
        a[6]  = (_Float16)a1.z; a[7]  = (_Float16)a1.w;
        a[8]  = (_Float16)a2.x; a[9]  = (_Float16)a2.y;
        a[10] = (_Float16)a2.z; a[11] = (_Float16)a2.w;
        a[12] = (_Float16)a3.x; a[13] = (_Float16)a3.y;
        a[14] = (_Float16)a3.z; a[15] = (_Float16)a3.w;
        // B 32x16 f16, same contiguous k order as A: lanes 0-15 -> K 0..15, lanes 16-31 -> K 16..31
#pragma unroll
        for (int j = 0; j < 16; ++j)
            b[j] = (_Float16)wcol[(size_t)(k0 + j) * NOUT];
        c = __builtin_amdgcn_wmma_f32_16x16x32_f16(false, a, false, b,
                                                   (short)0, c, false, false);
    }

    const float bb = bias[col];
    float* __restrict__ cp = C + (size_t)(rowTile + (hi << 3)) * NOUT + col;
    if (HAS_RESID) {
        const float* __restrict__ rp = resid + (size_t)(rowTile + (hi << 3)) * NOUT + col;
#pragma unroll
        for (int r = 0; r < 8; ++r)         // C/D: vgpr r -> row r (+8 for hi half)
            cp[(size_t)r * NOUT] = c[r] + bb + rp[(size_t)r * NOUT];
    } else {
#pragma unroll
        for (int r = 0; r < 8; ++r)
            cp[(size_t)r * NOUT] = c[r] + bb;
    }
}

// ---------------------------------------------------------------------------
// relu(LayerNorm(row)) — one wave32 per row, C in {64,128}.
// ---------------------------------------------------------------------------
__global__ void ln_relu_kernel(const float* __restrict__ in,
                               const float* __restrict__ g,
                               const float* __restrict__ b,
                               float* __restrict__ out, int M, int C)
{
    const int row  = blockIdx.x * (blockDim.x >> 5) + (threadIdx.x >> 5);
    const int lane = threadIdx.x & 31;
    if (row >= M) return;
    const int per = C >> 5;                 // 2 or 4 elements per lane
    float v[4];
    float sum = 0.0f;
    const float* rp = in + (size_t)row * C;
    for (int j = 0; j < per; ++j) { v[j] = rp[lane + (j << 5)]; sum += v[j]; }
    for (int off = 16; off; off >>= 1) sum += __shfl_xor(sum, off, 32);
    const float mu = sum / (float)C;
    float vs = 0.0f;
    for (int j = 0; j < per; ++j) { float d = v[j] - mu; vs += d * d; }
    for (int off = 16; off; off >>= 1) vs += __shfl_xor(vs, off, 32);
    const float inv = rsqrtf(vs / (float)C + LN_EPSF);
    float* op = out + (size_t)row * C;
    for (int j = 0; j < per; ++j) {
        const int ci = lane + (j << 5);
        op[ci] = fmaxf((v[j] - mu) * inv * g[ci] + b[ci], 0.0f);
    }
}

__global__ void init_scatter_kernel(float* __restrict__ smax,
                                    float* __restrict__ denom,
                                    float* __restrict__ num, int n)
{
    const int i = blockIdx.x * blockDim.x + threadIdx.x;
    if (i < n) { smax[i] = -3.402823466e38f; denom[i] = 0.0f; num[i] = 0.0f; }
}

// agg = num/denom (0 for empty segments) + conv input (GENConv root residual)
__global__ void agg_finalize_kernel(const float* __restrict__ num,
                                    const float* __restrict__ denom,
                                    const float* __restrict__ in,
                                    float* __restrict__ out, int n)
{
    const int i = blockIdx.x * blockDim.x + threadIdx.x;
    if (i < n) {
        const float dn = denom[i];
        out[i] = (dn > 0.0f ? num[i] / dn : 0.0f) + in[i];
    }
}

// ---------------------------------------------------------------------------
// Scatter passes: one wave32 per edge. Lanes 0-15 hold the 16 edge attrs,
// broadcast via __shfl; each lane produces features {lane, lane+32}. The
// edge encoding (attr @ edge_W + edge_b) is recomputed in-register each pass
// (51 MB of streaming reads vs 205 MB of materialized ea per pass).
// Node arrays (h, smax, denom, num: 12.8 MB each) are L2-resident.
// ---------------------------------------------------------------------------
__global__ void scatter_max_kernel(const float* __restrict__ h,
                                   const int* __restrict__ srcA,
                                   const int* __restrict__ dstA,
                                   const float* __restrict__ eattr,
                                   const float* __restrict__ edge_W,
                                   const float* __restrict__ edge_b,
                                   const float* __restrict__ tptr,
                                   int E, float* __restrict__ smax)
{
    const int e    = (blockIdx.x * blockDim.x + threadIdx.x) >> 5;
    const int lane = threadIdx.x & 31;
    if (e >= E) return;
    const int   s = srcA[e];
    const int   d = dstA[e];
    const float t = tptr[0];
    const float a = (lane < 16) ? eattr[(size_t)e * 16 + lane] : 0.0f;
    if (lane == 0) __builtin_prefetch(&eattr[(size_t)(e + 64) * 16], 0, 1);
#pragma unroll
    for (int p = 0; p < 2; ++p) {
        const int f = lane + (p << 5);
        float acc = edge_b[f];
#pragma unroll
        for (int k = 0; k < 16; ++k)
            acc = fmaf(__shfl(a, k, 32), edge_W[(k << 6) + f], acc);
        const float msg = fmaxf(h[(size_t)s * HDIM + f] + acc, 0.0f) + EPS_MSG;
        atomicMaxF(&smax[(size_t)d * HDIM + f], msg * t);
    }
}

__global__ void scatter_sum_kernel(const float* __restrict__ h,
                                   const int* __restrict__ srcA,
                                   const int* __restrict__ dstA,
                                   const float* __restrict__ eattr,
                                   const float* __restrict__ edge_W,
                                   const float* __restrict__ edge_b,
                                   const float* __restrict__ tptr,
                                   int E,
                                   const float* __restrict__ smax,
                                   float* __restrict__ denom,
                                   float* __restrict__ num)
{
    const int e    = (blockIdx.x * blockDim.x + threadIdx.x) >> 5;
    const int lane = threadIdx.x & 31;
    if (e >= E) return;
    const int   s = srcA[e];
    const int   d = dstA[e];
    const float t = tptr[0];
    const float a = (lane < 16) ? eattr[(size_t)e * 16 + lane] : 0.0f;
    if (lane == 0) __builtin_prefetch(&eattr[(size_t)(e + 64) * 16], 0, 1);
#pragma unroll
    for (int p = 0; p < 2; ++p) {
        const int f = lane + (p << 5);
        float acc = edge_b[f];
#pragma unroll
        for (int k = 0; k < 16; ++k)
            acc = fmaf(__shfl(a, k, 32), edge_W[(k << 6) + f], acc);
        const float msg = fmaxf(h[(size_t)s * HDIM + f] + acc, 0.0f) + EPS_MSG;
        const float sv  = msg * t;
        const float ev  = __expf(sv - smax[(size_t)d * HDIM + f]);
        atomicAdd(&denom[(size_t)d * HDIM + f], ev);
        atomicAdd(&num[(size_t)d * HDIM + f], msg * ev);
    }
}

// ---------------------------------------------------------------------------
extern "C" void kernel_launch(void* const* d_in, const int* in_sizes, int n_in,
                              void* d_out, int out_size, void* d_ws, size_t ws_size,
                              hipStream_t stream)
{
    const float* x      = (const float*)d_in[0];
    const int*   ei     = (const int*)  d_in[1];
    const float* eattr  = (const float*)d_in[2];
    const float* node_W = (const float*)d_in[3];
    const float* node_b = (const float*)d_in[4];
    const float* edge_W = (const float*)d_in[5];
    const float* edge_b = (const float*)d_in[6];
    const float* tparam = (const float*)d_in[7];
    const float* mlp1_W = (const float*)d_in[8];
    const float* mlp1_b = (const float*)d_in[9];
    const float* ln_g   = (const float*)d_in[10];
    const float* ln_b   = (const float*)d_in[11];
    const float* mlp2_W = (const float*)d_in[12];
    const float* mlp2_b = (const float*)d_in[13];
    const float* blk_g  = (const float*)d_in[14];
    const float* blk_b  = (const float*)d_in[15];

    const int N = in_sizes[0] / 128;      // 50000 (== 3125 * 16)
    const int E = in_sizes[1] / 2;        // 800000
    const int* src = ei;
    const int* dst = ei + E;
    const size_t NH = (size_t)N * HDIM;

    float* h      = (float*)d_ws;         // [N,64]
    float* z      = h      + NH;          // [N,64] pre-normed conv input
    float* aggout = z      + NH;          // [N,64] agg + root residual
    float* smax   = aggout + NH;          // [N,64]
    float* denom  = smax   + NH;          // [N,64]
    float* num    = denom  + NH;          // [N,64]
    float* mid    = num    + NH;          // [N,128] MLP hidden

    const dim3 blk128(128), blk256(256);
    const int rowTiles = N / 16;                    // 3125
    const int lnGrid   = (N + 7) / 8;               // 8 rows / 8 waves per block
    const int ewGrid   = (E + 7) / 8;               // 8 edges / 8 waves per block
    const int elGrid   = (int)((NH + 255) / 256);

    // Node encoder: h = x @ node_W + node_b   (K=128, NOUT=64)
    wmma_gemm_kernel<128, 64, false><<<dim3(rowTiles, 1), blk128, 0, stream>>>(
        x, node_W, node_b, nullptr, h);

    for (int i = 0; i < 3; ++i) {
        const float* convin = h;
        if (i > 0) {  // res+ pre-norm: z = relu(LN(h))
            ln_relu_kernel<<<lnGrid, blk256, 0, stream>>>(
                h, blk_g + i * 64, blk_b + i * 64, z, N, 64);
            convin = z;
        }
        init_scatter_kernel<<<elGrid, blk256, 0, stream>>>(smax, denom, num, (int)NH);
        scatter_max_kernel<<<ewGrid, blk256, 0, stream>>>(
            convin, src, dst, eattr, edge_W, edge_b, tparam + i, E, smax);
        scatter_sum_kernel<<<ewGrid, blk256, 0, stream>>>(
            convin, src, dst, eattr, edge_W, edge_b, tparam + i, E,
            smax, denom, num);
        agg_finalize_kernel<<<elGrid, blk256, 0, stream>>>(
            num, denom, convin, aggout, (int)NH);
        // mid = aggout @ W1 + b1   (K=64, NOUT=128)
        wmma_gemm_kernel<64, 128, false><<<dim3(rowTiles, 2), blk128, 0, stream>>>(
            aggout, mlp1_W + (size_t)i * 64 * 128, mlp1_b + i * 128, nullptr, mid);
        // mid = relu(LN(mid)) in place
        ln_relu_kernel<<<lnGrid, blk256, 0, stream>>>(
            mid, ln_g + i * 128, ln_b + i * 128, mid, N, 128);
        // h = mid @ W2 + b2 (+ h for res+ layers)   (K=128, NOUT=64)
        if (i == 0) {
            wmma_gemm_kernel<128, 64, false><<<dim3(rowTiles, 1), blk128, 0, stream>>>(
                mid, mlp2_W + (size_t)i * 128 * 64, mlp2_b + i * 64, nullptr, h);
        } else {
            wmma_gemm_kernel<128, 64, true><<<dim3(rowTiles, 1), blk128, 0, stream>>>(
                mid, mlp2_W + (size_t)i * 128 * 64, mlp2_b + i * 64, h, h);
        }
    }
    // Final: relu(LN(h, blk_ln[0]))
    ln_relu_kernel<<<lnGrid, blk256, 0, stream>>>(
        h, blk_g, blk_b, (float*)d_out, N, 64);
}
